// TransformerBlock_36696200577074
// MI455X (gfx1250) — compile-verified, matrix operand order checked by reading, not run on hardware
//
#include <hip/hip_runtime.h>
#include <hip/hip_bf16.h>

typedef __bf16 bf16_t;
typedef __attribute__((ext_vector_type(16))) __bf16 v16bf;
typedef __attribute__((ext_vector_type(8)))  __bf16 v8bf;
typedef __attribute__((ext_vector_type(8)))  float  v8f;
typedef __attribute__((ext_vector_type(4)))  float  v4f;

// ---------- helpers ----------
__device__ __forceinline__ bf16_t f2bf(float f) {
    union { float f; unsigned u; } a; a.f = f;
    unsigned r = a.u + 0x7fffu + ((a.u >> 16) & 1u);   // round-to-nearest-even
    union { unsigned short s; bf16_t b; } o; o.s = (unsigned short)(r >> 16);
    return o.b;
}

__device__ __forceinline__ v16bf cat8(v8bf lo, v8bf hi) {
    return __builtin_shufflevector(lo, hi, 0, 1, 2, 3, 4, 5, 6, 7,
                                   8, 9, 10, 11, 12, 13, 14, 15);
}

__device__ __forceinline__ v8f zero8() {
    v8f r;
#pragma unroll
    for (int i = 0; i < 8; ++i) r[i] = 0.0f;
    return r;
}

// ---------- f32 -> bf16 convert ----------
__global__ void cvt_f32_bf16_kernel(const float* __restrict__ s, bf16_t* __restrict__ d, int n) {
    int i = blockIdx.x * blockDim.x + threadIdx.x;
    if (i < n) d[i] = f2bf(s[i]);
}

// ---------- LayerNorm (DIM=512) -> bf16 ----------
__global__ __launch_bounds__(256) void ln_bf16_kernel(const float* __restrict__ x,
                                                      const float* __restrict__ g,
                                                      const float* __restrict__ b,
                                                      bf16_t* __restrict__ out) {
    const int row = blockIdx.x;
    const float* xr = x + (size_t)row * 512;
    __shared__ float red[256];
    const int t = threadIdx.x;
    float v0 = xr[t], v1 = xr[t + 256];
    red[t] = v0 + v1;
    __syncthreads();
#pragma unroll
    for (int off = 128; off > 0; off >>= 1) {
        if (t < off) red[t] += red[t + off];
        __syncthreads();
    }
    float mu = red[0] * (1.0f / 512.0f);
    __syncthreads();
    float d0 = v0 - mu, d1 = v1 - mu;
    red[t] = d0 * d0 + d1 * d1;
    __syncthreads();
#pragma unroll
    for (int off = 128; off > 0; off >>= 1) {
        if (t < off) red[t] += red[t + off];
        __syncthreads();
    }
    float rstd = rsqrtf(red[0] * (1.0f / 512.0f) + 1e-5f);
    bf16_t* orow = out + (size_t)row * 512;
    orow[t]       = f2bf(d0 * rstd * g[t] + b[t]);
    orow[t + 256] = f2bf(d1 * rstd * g[t + 256] + b[t + 256]);
}

// ---------- Generic WMMA GEMM:  C[M,N] = A[M,K] @ W[N,K]^T + bias ----------
// MODE 0: QKV scatter (Q pre-scaled by 1/sqrt(d)) to Q/K ([bh][tok][64]) and V^T ([bh][64][tok])
// MODE 1: residual f32 out:  outF = resid + C
// MODE 2: exact GELU, bf16 out
template <int MODE>
__global__ __launch_bounds__(256) void gemm_wmma_kernel(
    const bf16_t* __restrict__ A, const bf16_t* __restrict__ W,
    const float* __restrict__ bias, const float* __restrict__ resid,
    float* __restrict__ outF, bf16_t* __restrict__ outB,
    bf16_t* __restrict__ outQ, bf16_t* __restrict__ outK, bf16_t* __restrict__ outV,
    int M, int N, int K) {
    (void)M;
    __shared__ alignas(16) bf16_t ldsB[64 * 64];          // 64 cols x 64 k (bf16) = 8 KB

    const int lane = threadIdx.x & 31;
    const int w    = threadIdx.x >> 5;
    const int row  = lane & 15;
    const int hi   = lane >> 4;
    const int kb   = hi * 8;
    const size_t m0 = (size_t)blockIdx.y * 128 + (size_t)w * 16;
    const size_t n0 = (size_t)blockIdx.x * 64;

    v8f acc[4];
#pragma unroll
    for (int i = 0; i < 4; ++i) acc[i] = zero8();

    const int tbRow = threadIdx.x >> 2;                   // 0..63
    const int tbK   = (threadIdx.x & 3) * 16;             // 0,16,32,48

    for (int k0 = 0; k0 < K; k0 += 64) {
        // cooperative stage of B tile (rows of W are already the B-operand layout)
        const bf16_t* wp = W + (n0 + (size_t)tbRow) * (size_t)K + k0 + tbK;
        *(v8bf*)(&ldsB[tbRow * 64 + tbK])     = *(const v8bf*)(wp);
        *(v8bf*)(&ldsB[tbRow * 64 + tbK + 8]) = *(const v8bf*)(wp + 8);
        __syncthreads();

        if (k0 + 64 < K)
            __builtin_prefetch(W + (n0 + (size_t)tbRow) * (size_t)K + k0 + 64 + tbK, 0, 0);

#pragma unroll
        for (int kh = 0; kh < 64; kh += 32) {
            // A fragment: per-lane ISA layout (row = lane%16, K chunks kb and kb+16)
            const bf16_t* ar = A + (m0 + row) * (size_t)K + k0 + kh;
            v16bf af = cat8(*(const v8bf*)(ar + kb), *(const v8bf*)(ar + kb + 16));
#pragma unroll
            for (int nt = 0; nt < 4; ++nt) {
                const bf16_t* br = &ldsB[(nt * 16 + row) * 64 + kh];
                v16bf bf = cat8(*(const v8bf*)(br + kb), *(const v8bf*)(br + kb + 16));
                acc[nt] = __builtin_amdgcn_wmma_f32_16x16x32_bf16(
                    false, af, false, bf, (short)0, acc[nt], false, false);
            }
        }
        __syncthreads();
    }

    // epilogue: lane holds col n = n0 + nt*16 + row, rows m = m0 + r + 8*hi
#pragma unroll
    for (int nt = 0; nt < 4; ++nt) {
#pragma unroll
        for (int r = 0; r < 8; ++r) {
            size_t gm = m0 + r + hi * 8;
            int n = (int)n0 + nt * 16 + row;
            float v = acc[nt][r] + bias[n];
            if (MODE == 1) {
                outF[gm * (size_t)N + n] = resid[gm * (size_t)N + n] + v;
            } else if (MODE == 2) {
                float ge = 0.5f * v * (1.0f + erff(v * 0.70710678118654752f));
                outB[gm * (size_t)N + n] = f2bf(ge);
            } else {
                int comp = n >> 9, hh = (n >> 6) & 7, d = n & 63;
                int bb = (int)(gm >> 12), tok = (int)(gm & 4095);
                size_t bh = (size_t)(bb * 8 + hh);
                if (comp == 0)      outQ[(bh * 4096 + tok) * 64 + d] = f2bf(v * 0.125f);
                else if (comp == 1) outK[(bh * 4096 + tok) * 64 + d] = f2bf(v);
                else                outV[(bh * 64 + d) * 4096 + tok] = f2bf(v);
            }
        }
    }
}

// ---------- Flash attention: per wave = 16 query rows, stream 32 keys/iter ----------
// Softmax state is kept in A-layout association (row = lane%16, both half-waves
// agree after one shfl_xor(16)); S goes through one f32 LDS transpose so the
// exp'd P tile lands directly in A-operand registers for the P x V^T WMMA.
__global__ __launch_bounds__(256) void attn_kernel(const bf16_t* __restrict__ Q,
                                                   const bf16_t* __restrict__ Km,
                                                   const bf16_t* __restrict__ Vt,
                                                   bf16_t* __restrict__ O) {
    const int lane = threadIdx.x & 31;
    const int w    = threadIdx.x >> 5;
    const int row  = lane & 15;
    const int hi   = lane >> 4;
    const int kb   = hi * 8;
    const int bh   = blockIdx.y;                 // 0..15 (b*8+h)
    const int m0   = blockIdx.x * 128 + w * 16;  // query tile base (token in 0..4095)

    const bf16_t* qp = Q  + (size_t)bh * 4096 * 64;
    const bf16_t* kp = Km + (size_t)bh * 4096 * 64;
    const bf16_t* vp = Vt + (size_t)bh * 64 * 4096;

    __shared__ alignas(16) float ldsS[8][16 * 32];   // per-wave 16x32 f32 S tile (16 KB)

    // Q fragments for d=0..31 and d=32..63 (A-operand layout, Q pre-scaled by 1/8)
    const bf16_t* qr = qp + (size_t)(m0 + row) * 64;
    v16bf qa0 = cat8(*(const v8bf*)(qr + kb),      *(const v8bf*)(qr + kb + 16));
    v16bf qa1 = cat8(*(const v8bf*)(qr + 32 + kb), *(const v8bf*)(qr + 32 + kb + 16));

    v8f o[4];
#pragma unroll
    for (int i = 0; i < 4; ++i) o[i] = zero8();
    float rmA = -3.0e38f;   // running max for row = lane%16
    float rsA = 0.0f;       // running sum for row = lane%16

    for (int kc = 0; kc < 4096; kc += 32) {
        v8f s0 = zero8(), s1 = zero8();
        {   // keys kc .. kc+15
            const bf16_t* kr = kp + (size_t)(kc + row) * 64;
            v16bf b0 = cat8(*(const v8bf*)(kr + kb),      *(const v8bf*)(kr + kb + 16));
            v16bf b1 = cat8(*(const v8bf*)(kr + 32 + kb), *(const v8bf*)(kr + 32 + kb + 16));
            s0 = __builtin_amdgcn_wmma_f32_16x16x32_bf16(false, qa0, false, b0, (short)0, s0, false, false);
            s0 = __builtin_amdgcn_wmma_f32_16x16x32_bf16(false, qa1, false, b1, (short)0, s0, false, false);
        }
        {   // keys kc+16 .. kc+31
            const bf16_t* kr = kp + (size_t)(kc + 16 + row) * 64;
            v16bf b0 = cat8(*(const v8bf*)(kr + kb),      *(const v8bf*)(kr + kb + 16));
            v16bf b1 = cat8(*(const v8bf*)(kr + 32 + kb), *(const v8bf*)(kr + 32 + kb + 16));
            s1 = __builtin_amdgcn_wmma_f32_16x16x32_bf16(false, qa0, false, b0, (short)0, s1, false, false);
            s1 = __builtin_amdgcn_wmma_f32_16x16x32_bf16(false, qa1, false, b1, (short)0, s1, false, false);
        }

        // D-layout -> A-layout transpose of S via LDS (f32)
#pragma unroll
        for (int r = 0; r < 8; ++r) {
            int m = r + hi * 8;
            ldsS[w][m * 32 + row]      = s0[r];
            ldsS[w][m * 32 + 16 + row] = s1[r];
        }
        asm volatile("s_wait_dscnt 0" ::: "memory");

        // this lane now owns 16 of row(lane%16)'s 32 scores; other half on lane^16
        const float* sr = &ldsS[w][row * 32 + kb];
        v4f sv0 = *(const v4f*)(sr);
        v4f sv1 = *(const v4f*)(sr + 4);
        v4f sv2 = *(const v4f*)(sr + 16);
        v4f sv3 = *(const v4f*)(sr + 20);

        float mx = sv0[0];
#pragma unroll
        for (int i = 1; i < 4; ++i) mx = fmaxf(mx, sv0[i]);
#pragma unroll
        for (int i = 0; i < 4; ++i) mx = fmaxf(mx, sv1[i]);
#pragma unroll
        for (int i = 0; i < 4; ++i) mx = fmaxf(mx, sv2[i]);
#pragma unroll
        for (int i = 0; i < 4; ++i) mx = fmaxf(mx, sv3[i]);
        mx = fmaxf(mx, __shfl_xor(mx, 16, 32));          // merge key halves

        float nm = fmaxf(rmA, mx);
        float fA = __expf(rmA - nm);
        rmA = nm;

        float p[16];
#pragma unroll
        for (int i = 0; i < 4; ++i) {
            p[i]      = __expf(sv0[i] - nm);
            p[i + 4]  = __expf(sv1[i] - nm);
            p[i + 8]  = __expf(sv2[i] - nm);
            p[i + 12] = __expf(sv3[i] - nm);
        }
        float ps = 0.0f;
#pragma unroll
        for (int i = 0; i < 16; ++i) ps += p[i];
        ps += __shfl_xor(ps, 16, 32);
        rsA = rsA * fA + ps;

        // P tile is already in A-operand layout
        v16bf pf;
#pragma unroll
        for (int i = 0; i < 16; ++i) pf[i] = f2bf(p[i]);

        // rescale accumulators: broadcast fA from A-layout row owner to D-layout rows
#pragma unroll
        for (int r = 0; r < 8; ++r) {
            float fD = __shfl(fA, r + 8 * hi, 32);
#pragma unroll
            for (int t = 0; t < 4; ++t) o[t][r] *= fD;
        }

#pragma unroll
        for (int dt = 0; dt < 4; ++dt) {
            const bf16_t* vr = vp + (size_t)(dt * 16 + row) * 4096 + kc;
            v16bf bv = cat8(*(const v8bf*)(vr + kb), *(const v8bf*)(vr + kb + 16));
            o[dt] = __builtin_amdgcn_wmma_f32_16x16x32_bf16(
                false, pf, false, bv, (short)0, o[dt], false, false);
        }
    }

    const int bb = bh >> 3, hh = bh & 7;
#pragma unroll
    for (int r = 0; r < 8; ++r) {
        float rsD = __shfl(rsA, r + 8 * hi, 32);
        float inv = 1.0f / rsD;
        int tok = m0 + r + hi * 8;
        bf16_t* orow = O + ((size_t)(bb * 4096 + tok)) * 512 + hh * 64 + row;
#pragma unroll
        for (int dt = 0; dt < 4; ++dt)
            orow[dt * 16] = f2bf(o[dt][r] * inv);
    }
}

// ---------- workspace layout (bf16 element offsets unless noted) ----------
static constexpr size_t E_WQKV  = 0;                       // 1536*512
static constexpr size_t E_WPROJ = E_WQKV  + 1536 * 512;    // 512*512
static constexpr size_t E_WFC1  = E_WPROJ + 512 * 512;     // 2048*512
static constexpr size_t E_WFC2  = E_WFC1  + 2048 * 512;    // 512*2048
static constexpr size_t E_H     = E_WFC2  + 512 * 2048;    // 8192*512 (LN out, reused)
static constexpr size_t E_Q     = E_H     + 8192 * 512;    // 16*4096*64
static constexpr size_t E_K     = E_Q     + 16 * 4096 * 64;
static constexpr size_t E_V     = E_K     + 16 * 4096 * 64;
static constexpr size_t E_ATTN  = E_V     + 16 * 4096 * 64; // 8192*512
static constexpr size_t B_X1    = (E_ATTN + 8192 * 512) * 2; // byte offset, f32 8192*512
static constexpr size_t E_HID   = E_Q;                     // alias Q..ATTN (8192*2048 bf16)

extern "C" void kernel_launch(void* const* d_in, const int* in_sizes, int n_in,
                              void* d_out, int out_size, void* d_ws, size_t ws_size,
                              hipStream_t stream) {
    (void)in_sizes; (void)n_in; (void)out_size; (void)ws_size;
    const float* x      = (const float*)d_in[0];
    const float* ln1_g  = (const float*)d_in[1];
    const float* ln1_b  = (const float*)d_in[2];
    const float* ln2_g  = (const float*)d_in[3];
    const float* ln2_b  = (const float*)d_in[4];
    const float* qkv_w  = (const float*)d_in[5];
    const float* qkv_b  = (const float*)d_in[6];
    const float* proj_w = (const float*)d_in[7];
    const float* proj_b = (const float*)d_in[8];
    const float* fc1_w  = (const float*)d_in[9];
    const float* fc1_b  = (const float*)d_in[10];
    const float* fc2_w  = (const float*)d_in[11];
    const float* fc2_b  = (const float*)d_in[12];

    bf16_t* wsb   = (bf16_t*)d_ws;
    bf16_t* Wqkv  = wsb + E_WQKV;
    bf16_t* Wproj = wsb + E_WPROJ;
    bf16_t* Wfc1  = wsb + E_WFC1;
    bf16_t* Wfc2  = wsb + E_WFC2;
    bf16_t* bufH  = wsb + E_H;
    bf16_t* bufQ  = wsb + E_Q;
    bf16_t* bufK  = wsb + E_K;
    bf16_t* bufV  = wsb + E_V;
    bf16_t* bufA  = wsb + E_ATTN;
    bf16_t* bufHid= wsb + E_HID;
    float*  bufX1 = (float*)((char*)d_ws + B_X1);
    float*  out   = (float*)d_out;

    // 1) weight conversions to bf16
    cvt_f32_bf16_kernel<<<(1536 * 512 + 255) / 256, 256, 0, stream>>>(qkv_w,  Wqkv,  1536 * 512);
    cvt_f32_bf16_kernel<<<(512  * 512 + 255) / 256, 256, 0, stream>>>(proj_w, Wproj, 512 * 512);
    cvt_f32_bf16_kernel<<<(2048 * 512 + 255) / 256, 256, 0, stream>>>(fc1_w,  Wfc1,  2048 * 512);
    cvt_f32_bf16_kernel<<<(512 * 2048 + 255) / 256, 256, 0, stream>>>(fc2_w,  Wfc2,  512 * 2048);

    // 2) LN1
    ln_bf16_kernel<<<8192, 256, 0, stream>>>(x, ln1_g, ln1_b, bufH);

    // 3) QKV projection + scatter (M=8192, N=1536, K=512)
    gemm_wmma_kernel<0><<<dim3(24, 64), 256, 0, stream>>>(
        bufH, Wqkv, qkv_b, nullptr, nullptr, nullptr, bufQ, bufK, bufV, 8192, 1536, 512);

    // 4) attention (16 bh-slices x 32 row-tiles of 128)
    attn_kernel<<<dim3(32, 16), 256, 0, stream>>>(bufQ, bufK, bufV, bufA);

    // 5) output projection + residual (f32)
    gemm_wmma_kernel<1><<<dim3(8, 64), 256, 0, stream>>>(
        bufA, Wproj, proj_b, x, bufX1, nullptr, nullptr, nullptr, nullptr, 8192, 512, 512);

    // 6) LN2
    ln_bf16_kernel<<<8192, 256, 0, stream>>>(bufX1, ln2_g, ln2_b, bufH);

    // 7) FC1 + exact GELU -> bf16 hidden (N=2048)
    gemm_wmma_kernel<2><<<dim3(32, 64), 256, 0, stream>>>(
        bufH, Wfc1, fc1_b, nullptr, nullptr, bufHid, nullptr, nullptr, nullptr, 8192, 2048, 512);

    // 8) FC2 + residual -> f32 output (K=2048)
    gemm_wmma_kernel<1><<<dim3(8, 64), 256, 0, stream>>>(
        bufHid, Wfc2, fc2_b, bufX1, out, nullptr, nullptr, nullptr, nullptr, 8192, 512, 2048);
}
